// OptimizedMoELayer_40089224741098
// MI455X (gfx1250) — matrix operand order repaired and checked by reading, not compiled, not run on hardware
//
#include <hip/hip_runtime.h>
#include <hip/hip_bf16.h>

#define TTOK 2048
#define HDIM 1024
#define IDIM 4096
#define NEXP 8
#define NLOG 9   // experts + shared router logits

typedef __attribute__((ext_vector_type(16))) __bf16 v16bf;
typedef __attribute__((ext_vector_type(8)))  __bf16 bf8;
typedef __attribute__((ext_vector_type(8)))  float   v8f;

#define AS3 __attribute__((address_space(3)))

__device__ __forceinline__ float silu_f(float x) { return x / (1.0f + __expf(-x)); }

// 32-bit LDS byte offset of a __shared__ object (for async-to-LDS VDST operand)
__device__ __forceinline__ unsigned lds_addr_u32(const void* p) {
  return (unsigned)(size_t)(AS3 const void*)p;
}

// Per-lane async copy of 16B global -> LDS (tracked by ASYNCcnt)
__device__ __forceinline__ void async_copy_b128(unsigned lds_off, const void* gptr) {
  asm volatile("global_load_async_to_lds_b128 %0, %1, off"
               :: "v"(lds_off), "v"(gptr)
               : "memory");
}
__device__ __forceinline__ void wait_asynccnt0() {
  asm volatile("s_wait_asynccnt 0x0" ::: "memory");
}

// ---------------------------------------------------------------------------
// Router: logits -> softmax -> top-2 -> gather lists; also emits bf16 copy of x
// One wave (32 lanes) per token. Block = 128 threads = 4 tokens.
// ---------------------------------------------------------------------------
__global__ __launch_bounds__(128) void moe_router_k(
    const float* __restrict__ x, const float* __restrict__ rw,
    __bf16* __restrict__ xb, float* __restrict__ wlist,
    int* __restrict__ ilist, int* __restrict__ counts,
    float* __restrict__ zbuf) {
  const int tid  = threadIdx.x;
  const int lane = tid & 31;
  const int t    = blockIdx.x * 4 + (tid >> 5);

  float part[NLOG];
#pragma unroll
  for (int e = 0; e < NLOG; ++e) part[e] = 0.0f;

  const float* xr = x + (size_t)t * HDIM;
#pragma unroll 4
  for (int j = 0; j < HDIM / 32; ++j) {
    const int h = j * 32 + lane;
    const float xv = xr[h];
    xb[(size_t)t * HDIM + h] = (__bf16)xv;
#pragma unroll
    for (int e = 0; e < NLOG; ++e) part[e] += xv * rw[e * HDIM + h];
  }
  // wave32 butterfly reduction
#pragma unroll
  for (int off = 16; off > 0; off >>= 1) {
#pragma unroll
    for (int e = 0; e < NLOG; ++e) part[e] += __shfl_xor(part[e], off, 32);
  }

  if (lane == 0) {
    float mx = part[0];
#pragma unroll
    for (int e = 1; e < NLOG; ++e) mx = fmaxf(mx, part[e]);
    float p[NLOG], s = 0.0f, z = 0.0f;
#pragma unroll
    for (int e = 0; e < NLOG; ++e) {
      p[e] = __expf(part[e] - mx);
      s += p[e];
      z += part[e] * part[e];
    }
    const float inv = 1.0f / s;
#pragma unroll
    for (int e = 0; e < NLOG; ++e) p[e] *= inv;
    zbuf[t] = z;

    int i0 = 0;
#pragma unroll
    for (int e = 1; e < NLOG; ++e) if (p[e] > p[i0]) i0 = e;
    int i1 = (i0 == 0) ? 1 : 0;
#pragma unroll
    for (int e = 0; e < NLOG; ++e) if (e != i0 && p[e] > p[i1]) i1 = e;

    const float denom = p[i0] + p[i1] + 1e-6f;
    const float w0 = p[i0] / denom, w1 = p[i1] / denom;
    if (i0 < NEXP) {
      int pos = atomicAdd(&counts[i0], 1);
      ilist[i0 * TTOK + pos] = t;
      wlist[i0 * TTOK + pos] = w0;
    }
    if (i1 < NEXP) {
      int pos = atomicAdd(&counts[i1], 1);
      ilist[i1 * TTOK + pos] = t;
      wlist[i1 * TTOK + pos] = w1;
    }
  }
}

// ---------------------------------------------------------------------------
// GEMM1: act[i, n] = silu(x[tok_i] . gate[n]) * (x[tok_i] . up[n])
// Block tile M=64 (gathered tokens) x N=128 (I cols), K-step 32 over H.
// 256 threads = 8 waves, each wave owns a 16x64 subtile (4 wmma per matrix).
// A tiles: async global->LDS (bf16, no conversion). Weights: fp32->bf16 in LDS.
// ---------------------------------------------------------------------------
__global__ __launch_bounds__(256) void moe_gemm1_k(
    const __bf16* __restrict__ xb, const float* __restrict__ gw,
    const float* __restrict__ uw, const int* __restrict__ ilist,
    const int* __restrict__ cnt_ptr, __bf16* __restrict__ act) {
  const int n0 = blockIdx.x * 128;
  const int m0 = blockIdx.y * 64;
  const int cnt = cnt_ptr ? *cnt_ptr : TTOK;
  if (m0 >= cnt) return;

  __shared__ __bf16 As[64][40];    // row pad: 40 halves = 80B -> conflict-free
  __shared__ __bf16 Gs[128][40];
  __shared__ __bf16 Us[128][40];
  __shared__ int toks[64];

  const int tid = threadIdx.x;
  if (tid < 64) {
    const int r = m0 + tid;
    toks[tid] = (r < cnt) ? (ilist ? ilist[r] : r) : -1;
  }

  const int ar_ld = tid >> 2;            // A staging: 64 rows x 4 chunks of 8 bf16
  const int ac_ld = (tid & 3) * 8;
  const int br_ld = tid >> 1;            // B staging: 128 rows x 2 chunks of 16 f32
  const int bc_ld = (tid & 1) * 16;
  const unsigned a_lds = lds_addr_u32(&As[ar_ld][ac_ld]);

  __syncthreads();
  const int atok = toks[ar_ld];
  if (atok < 0) {                        // zero rows beyond cnt, once; async lanes
    const bf8 z = {};                    // for these rows stay EXEC-masked off
    *(bf8*)&As[ar_ld][ac_ld] = z;
  }

  const int lane = tid & 31;
  const int wav  = tid >> 5;
  const int mw   = wav & 3;    // 4 M-subtiles of 16
  const int nw   = wav >> 2;   // 2 N-subtiles of 64

  v8f accg[4], accu[4];
  const v8f vzero = {};
#pragma unroll
  for (int j = 0; j < 4; ++j) { accg[j] = vzero; accu[j] = vzero; }

  for (int k0 = 0; k0 < HDIM; k0 += 32) {
    // ---- weight fetch into VGPRs (overlaps previous compute) ----
    const float* gp = gw + (size_t)(n0 + br_ld) * HDIM + k0 + bc_ld;
    const float* up = uw + (size_t)(n0 + br_ld) * HDIM + k0 + bc_ld;
    float gv[16], uv[16];
#pragma unroll
    for (int q = 0; q < 4; ++q) {
      const float4 a = ((const float4*)gp)[q];
      gv[4 * q + 0] = a.x; gv[4 * q + 1] = a.y; gv[4 * q + 2] = a.z; gv[4 * q + 3] = a.w;
      const float4 b = ((const float4*)up)[q];
      uv[4 * q + 0] = b.x; uv[4 * q + 1] = b.y; uv[4 * q + 2] = b.z; uv[4 * q + 3] = b.w;
    }
    if (k0 + 32 < HDIM) {
      __builtin_prefetch(gp + 32, 0, 0);
      __builtin_prefetch(up + 32, 0, 0);
    }

    __syncthreads();   // previous iteration's LDS reads done
    // ---- A tile: async global->LDS, overlapped with weight conversion ----
    if (atok >= 0) {
      async_copy_b128(a_lds, xb + (size_t)atok * HDIM + k0 + ac_ld);
    }
    bf8 gA, gB, uA, uB;
#pragma unroll
    for (int q = 0; q < 8; ++q) {
      gA[q] = (__bf16)gv[q];     gB[q] = (__bf16)gv[8 + q];
      uA[q] = (__bf16)uv[q];     uB[q] = (__bf16)uv[8 + q];
    }
    *(bf8*)&Gs[br_ld][bc_ld]     = gA;
    *(bf8*)&Gs[br_ld][bc_ld + 8] = gB;
    *(bf8*)&Us[br_ld][bc_ld]     = uA;
    *(bf8*)&Us[br_ld][bc_ld + 8] = uB;
    wait_asynccnt0();
    __syncthreads();

    // ---- fragments: contiguous k order (k = 16*half + s), same for A and B ----
    // A 16x32: lanes 0-15 -> K 0..15 ; lanes 16-31 -> K 16..31
    const int arow = mw * 16 + (lane & 15);
    const int kbA  = (lane < 16) ? 0 : 16;
    const bf8 alo = *(const bf8*)&As[arow][kbA];
    const bf8 ahi = *(const bf8*)&As[arow][kbA + 8];
    v16bf afrag;
#pragma unroll
    for (int q = 0; q < 8; ++q) { afrag[q] = alo[q]; afrag[8 + q] = ahi[q]; }

    // B 32x16: lanes 0-15 -> K 0..15, N=lane ; lanes 16-31 -> K 16..31, N=lane-16
    const int kbB = (lane < 16) ? 0 : 16;
#pragma unroll
    for (int j = 0; j < 4; ++j) {
      const int bn = nw * 64 + j * 16 + (lane & 15);
      const bf8 b0 = *(const bf8*)&Gs[bn][kbB];
      const bf8 b1 = *(const bf8*)&Gs[bn][kbB + 8];
      v16bf bfr;
#pragma unroll
      for (int q = 0; q < 8; ++q) { bfr[q] = b0[q]; bfr[8 + q] = b1[q]; }
      accg[j] = __builtin_amdgcn_wmma_f32_16x16x32_bf16(
          false, afrag, false, bfr, (short)0, accg[j], false, false);

      const bf8 c0 = *(const bf8*)&Us[bn][kbB];
      const bf8 c1 = *(const bf8*)&Us[bn][kbB + 8];
      v16bf ufr;
#pragma unroll
      for (int q = 0; q < 8; ++q) { ufr[q] = c0[q]; ufr[8 + q] = c1[q]; }
      accu[j] = __builtin_amdgcn_wmma_f32_16x16x32_bf16(
          false, afrag, false, ufr, (short)0, accu[j], false, false);
    }
  }

  // ---- epilogue: act = silu(g) * u, bf16, compacted by gathered row ----
#pragma unroll
  for (int j = 0; j < 4; ++j) {
    const int col = n0 + nw * 64 + j * 16 + (lane & 15);
#pragma unroll
    for (int i = 0; i < 8; ++i) {
      const int rb = mw * 16 + i + ((lane < 16) ? 0 : 8);
      const int gi = m0 + rb;
      if (gi < cnt) {
        const float g = accg[j][i];
        const float u = accu[j][i];
        act[(size_t)gi * IDIM + col] = (__bf16)(silu_f(g) * u);
      }
    }
  }
}

// ---------------------------------------------------------------------------
// GEMM2: out[tok_i, h] += w_i * (act[i,:] . down[h,:]),  K over I=4096
// ---------------------------------------------------------------------------
__global__ __launch_bounds__(256) void moe_gemm2_k(
    const __bf16* __restrict__ act, const float* __restrict__ dwn,
    const int* __restrict__ ilist, const float* __restrict__ wlist,
    const int* __restrict__ cnt_ptr, float* __restrict__ out) {
  const int n0 = blockIdx.x * 128;   // H columns
  const int m0 = blockIdx.y * 64;
  const int cnt = cnt_ptr ? *cnt_ptr : TTOK;
  if (m0 >= cnt) return;

  __shared__ __bf16 As[64][40];
  __shared__ __bf16 Bs[128][40];
  __shared__ int   toks[64];
  __shared__ float wts[64];

  const int tid = threadIdx.x;
  if (tid < 64) {
    const int r = m0 + tid;
    toks[tid] = (r < cnt) ? (ilist ? ilist[r] : r) : 0;
    wts[tid]  = (r < cnt) ? (wlist ? wlist[r] : 1.0f) : 0.0f;
  }

  const int ar_ld = tid >> 2;
  const int ac_ld = (tid & 3) * 8;
  const int br_ld = tid >> 1;
  const int bc_ld = (tid & 1) * 16;
  const bool a_ok = (m0 + ar_ld) < cnt;
  const unsigned a_lds = lds_addr_u32(&As[ar_ld][ac_ld]);

  if (!a_ok) {                       // zero rows beyond cnt once
    const bf8 z = {};
    *(bf8*)&As[ar_ld][ac_ld] = z;
  }
  __syncthreads();

  const int lane = tid & 31;
  const int wav  = tid >> 5;
  const int mw   = wav & 3;
  const int nw   = wav >> 2;

  v8f acc[4];
  const v8f vzero = {};
#pragma unroll
  for (int j = 0; j < 4; ++j) acc[j] = vzero;

  for (int k0 = 0; k0 < IDIM; k0 += 32) {
    const float* bp = dwn + (size_t)(n0 + br_ld) * IDIM + k0 + bc_ld;
    float bv[16];
#pragma unroll
    for (int q = 0; q < 4; ++q) {
      const float4 a = ((const float4*)bp)[q];
      bv[4 * q + 0] = a.x; bv[4 * q + 1] = a.y; bv[4 * q + 2] = a.z; bv[4 * q + 3] = a.w;
    }
    if (k0 + 32 < IDIM) __builtin_prefetch(bp + 32, 0, 0);

    __syncthreads();
    if (a_ok) {
      async_copy_b128(a_lds, act + (size_t)(m0 + ar_ld) * IDIM + k0 + ac_ld);
    }
    bf8 bA, bB;
#pragma unroll
    for (int q = 0; q < 8; ++q) { bA[q] = (__bf16)bv[q]; bB[q] = (__bf16)bv[8 + q]; }
    *(bf8*)&Bs[br_ld][bc_ld]     = bA;
    *(bf8*)&Bs[br_ld][bc_ld + 8] = bB;
    wait_asynccnt0();
    __syncthreads();

    const int arow = mw * 16 + (lane & 15);
    const int kbA  = (lane < 16) ? 0 : 16;
    const bf8 alo = *(const bf8*)&As[arow][kbA];
    const bf8 ahi = *(const bf8*)&As[arow][kbA + 8];
    v16bf afrag;
#pragma unroll
    for (int q = 0; q < 8; ++q) { afrag[q] = alo[q]; afrag[8 + q] = ahi[q]; }

    const int kbB = (lane < 16) ? 0 : 16;
#pragma unroll
    for (int j = 0; j < 4; ++j) {
      const int bn = nw * 64 + j * 16 + (lane & 15);
      const bf8 b0 = *(const bf8*)&Bs[bn][kbB];
      const bf8 b1 = *(const bf8*)&Bs[bn][kbB + 8];
      v16bf bfr;
#pragma unroll
      for (int q = 0; q < 8; ++q) { bfr[q] = b0[q]; bfr[8 + q] = b1[q]; }
      acc[j] = __builtin_amdgcn_wmma_f32_16x16x32_bf16(
          false, afrag, false, bfr, (short)0, acc[j], false, false);
    }
  }

  // weighted scatter-add (experts serialized on stream -> plain RMW is safe;
  // (tok,col) pairs are unique within this kernel)
#pragma unroll
  for (int j = 0; j < 4; ++j) {
    const int col = n0 + nw * 64 + j * 16 + (lane & 15);
#pragma unroll
    for (int i = 0; i < 8; ++i) {
      const int rb = mw * 16 + i + ((lane < 16) ? 0 : 8);
      const int gi = m0 + rb;
      if (gi < cnt) {
        const int tok  = toks[rb];
        const float wt = wts[rb];
        float* p = out + (size_t)tok * HDIM + col;
        *p += wt * acc[j][i];
      }
    }
  }
}

// ---------------------------------------------------------------------------
// Aux losses (deterministic single-block reduction)
// ---------------------------------------------------------------------------
__global__ __launch_bounds__(256) void moe_loss_k(
    const float* __restrict__ zbuf, const int* __restrict__ counts,
    float* __restrict__ outp) {
  __shared__ float red[256];
  const int tid = threadIdx.x;
  float s = 0.0f;
  for (int i = tid; i < TTOK; i += 256) s += zbuf[i];
  red[tid] = s;
  __syncthreads();
  for (int off = 128; off > 0; off >>= 1) {
    if (tid < off) red[tid] += red[tid + off];
    __syncthreads();
  }
  if (tid == 0) {
    const float z = red[0] / (float)TTOK;
    float loads[NLOG];
    float tot = 0.0f;
    for (int e = 0; e < NEXP; ++e) { loads[e] = (float)counts[e]; tot += loads[e]; }
    loads[NEXP] = (float)TTOK; tot += loads[NEXP];
    const float ideal = 1.0f / (float)NLOG;
    float lb = 0.0f;
    for (int e = 0; e < NLOG; ++e) {
      const float d = loads[e] / (tot + 1e-6f) - ideal;
      lb += d * d;
    }
    lb /= (float)NLOG;
    outp[0] = 0.01f * lb + 0.01f * z;
    outp[1] = lb;
    outp[2] = z;
  }
}

// ---------------------------------------------------------------------------
extern "C" void kernel_launch(void* const* d_in, const int* in_sizes, int n_in,
                              void* d_out, int out_size, void* d_ws, size_t ws_size,
                              hipStream_t stream) {
  (void)in_sizes; (void)n_in; (void)out_size; (void)ws_size;
  const float* x   = (const float*)d_in[0];
  const float* rw  = (const float*)d_in[1];
  const float* gw  = (const float*)d_in[2];
  const float* uw  = (const float*)d_in[3];
  const float* dwn = (const float*)d_in[4];
  const float* sgw = (const float*)d_in[5];
  const float* suw = (const float*)d_in[6];
  const float* sdw = (const float*)d_in[7];
  float* out = (float*)d_out;

  char* w = (char*)d_ws;
  __bf16* xb   = (__bf16*)w;  w += (size_t)TTOK * HDIM * sizeof(__bf16);
  __bf16* act  = (__bf16*)w;  w += (size_t)TTOK * IDIM * sizeof(__bf16);
  float*  zbuf = (float*)w;   w += (size_t)TTOK * sizeof(float);
  float* wlist = (float*)w;   w += (size_t)NEXP * TTOK * sizeof(float);
  int*   ilist = (int*)w;     w += (size_t)NEXP * TTOK * sizeof(int);
  int*  counts = (int*)w;     w += 256;

  hipMemsetAsync(out, 0, (size_t)TTOK * HDIM * sizeof(float), stream);
  hipMemsetAsync(counts, 0, NEXP * sizeof(int), stream);

  moe_router_k<<<TTOK / 4, 128, 0, stream>>>(x, rw, xb, wlist, ilist, counts, zbuf);

  const dim3 g1(IDIM / 128, TTOK / 64);
  const dim3 g2(HDIM / 128, TTOK / 64);
  for (int e = 0; e < NEXP; ++e) {
    moe_gemm1_k<<<g1, 256, 0, stream>>>(xb, gw + (size_t)e * IDIM * HDIM,
                                        uw + (size_t)e * IDIM * HDIM,
                                        ilist + e * TTOK, counts + e, act);
    moe_gemm2_k<<<g2, 256, 0, stream>>>(act, dwn + (size_t)e * HDIM * IDIM,
                                        ilist + e * TTOK, wlist + e * TTOK,
                                        counts + e, out);
  }
  // Shared expert: identity token list, weight 1/SHARED = 1.0
  moe_gemm1_k<<<g1, 256, 0, stream>>>(xb, sgw, suw, nullptr, nullptr, act);
  moe_gemm2_k<<<g2, 256, 0, stream>>>(act, sdw, nullptr, nullptr, nullptr, out);

  moe_loss_k<<<1, 256, 0, stream>>>(zbuf, counts, out + (size_t)TTOK * HDIM);
}
